// FNOBlock_8126078124546
// MI455X (gfx1250) — matrix operand order repaired and checked by reading, not compiled, not run on hardware
//
#include <hip/hip_runtime.h>

// ---------------------------------------------------------------------------
// FNO block for MI455X (gfx1250, wave32, WMMA).
// Pipeline:
//   K1: 256-pt FFT over rows of x          (x real -> X complex, in ws)
//   K2: 512-pt FFT over columns            (X -> Xc, in ws)
//   K0: repack w1 / lin1_w^T / lin2_w^T to bf16 WMMA-B fragment layout (ws)
//   K3: fused  lift+crelu+phase -> GEMM(w1) -> GEMM(lin1^T)+b+crelu
//              -> GEMM(lin2^T)+b -> 256-pt IFFT over v  -> d_out
//   K4: in-place 256-pt IFFT over m on d_out
//   K5: in-place 512-pt IFFT over b on d_out
// GEMMs use v_wmma_f32_16x16x32_bf16 with f32 accumulation.
// ---------------------------------------------------------------------------

typedef __attribute__((ext_vector_type(16))) __bf16 vbf16;
typedef __attribute__((ext_vector_type(8)))  float  v8f;

#define LDA 264                    // padded bf16 row stride (264*2B = 528B, 16B aligned, bank-skewed)
#define TWO_PI 6.28318530717958647692f

__device__ __forceinline__ float2 cmulf(float2 a, float2 b) {
  return make_float2(a.x*b.x - a.y*b.y, a.x*b.y + a.y*b.x);
}

// --- block-cooperative radix-2 DIT stages (input already bit-reversed) -----
__device__ __forceinline__ void fft_stages_block(float2* s, int N, int logN,
                                                 float sign, int tid, int T) {
  for (int st = 1; st <= logN; ++st) {
    int len = 1 << st, half = len >> 1;
    for (int j = tid; j < (N >> 1); j += T) {
      int blk = j >> (st - 1);
      int off = j & (half - 1);
      int i0 = blk * len + off, i1 = i0 + half;
      float ang = sign * TWO_PI * (float)off / (float)len;
      float sn, cs; __sincosf(ang, &sn, &cs);
      float2 a = s[i0];
      float2 b = cmulf(s[i1], make_float2(cs, sn));
      s[i0] = make_float2(a.x + b.x, a.y + b.y);
      s[i1] = make_float2(a.x - b.x, a.y - b.y);
    }
    __syncthreads();
  }
}

// --- K1: forward FFT over each 256-wide row of real x ----------------------
__global__ void fno_fft_rows(const float* __restrict__ x, float2* __restrict__ X) {
  __shared__ float2 s[256];
  const int row = blockIdx.x;          // 512 rows
  const int tid = threadIdx.x;         // 128 threads
  for (int n = tid; n < 256; n += 128) {
    int rn = (int)(__brev((unsigned)n) >> 24);
    s[n] = make_float2(x[row * 256 + rn], 0.0f);
  }
  __syncthreads();
  fft_stages_block(s, 256, 8, -1.0f, tid, 128);
  for (int n = tid; n < 256; n += 128) X[row * 256 + n] = s[n];
}

// --- K2: forward FFT over each 512-long column -----------------------------
__global__ void fno_fft_cols(const float2* __restrict__ Xin, float2* __restrict__ Xout) {
  __shared__ float2 s[512];
  const int c = blockIdx.x;            // 256 columns
  const int tid = threadIdx.x;         // 256 threads
  for (int n = tid; n < 512; n += 256) {
    int rn = (int)(__brev((unsigned)n) >> 23);
    s[n] = Xin[rn * 256 + c];
  }
  __syncthreads();
  fft_stages_block(s, 512, 9, -1.0f, tid, 256);
  for (int n = tid; n < 512; n += 256) Xout[n * 256 + c] = s[n];
}

// --- K0: repack weights to bf16 WMMA B-fragment layout ---------------------
// B tile (kt,nt): 512 bf16 laid out so lane reads 16 contiguous bf16.
// k = kt*32 + 8*(lane>=16) + (e&7) + 16*(e>=8), n = nt*16 + (lane&15)   (ISA 7.12.2 B layout)
__global__ void fno_repack(const float* __restrict__ w1,
                           const float* __restrict__ l1,
                           const float* __restrict__ l2,
                           __bf16* __restrict__ w1p,
                           __bf16* __restrict__ l1p,
                           __bf16* __restrict__ l2p) {
  int g = blockIdx.x * blockDim.x + threadIdx.x;   // 3 * 65536
  if (g >= 3 * 65536) return;
  int which = g >> 16;
  int o     = g & 65535;
  int nt = o >> 12, rem = o & 4095;
  int kt = rem >> 9, q = rem & 511;
  int lane = q >> 4, e = q & 15;
  int k = kt * 32 + (e & 7) + ((e >= 8) ? 16 : 0) + ((lane >= 16) ? 8 : 0);
  int n = nt * 16 + (lane & 15);
  if (which == 0)      w1p[o] = (__bf16)w1[k * 256 + n];        // h @ w1
  else if (which == 1) l1p[o] = (__bf16)l1[n * 256 + k];        // h @ lin1_w^T
  else                 l2p[o] = (__bf16)l2[n * 256 + k];        // h @ lin2_w^T
}

// --- WMMA fragment loaders --------------------------------------------------
union FragBF { vbf16 v; uint4 q[2]; };

__device__ __forceinline__ vbf16 load_bfrag(const __bf16* __restrict__ Bp,
                                            int nt, int kt, int lane) {
  FragBF u;
  const uint4* p = (const uint4*)(Bp + (((nt << 3) + kt) << 9) + (lane << 4));
  u.q[0] = p[0];
  u.q[1] = p[1];
  return u.v;
}

// A 16x32 bf16 fragment from LDS tile (row-major, stride LDA):
// element e<8  -> k = kt*32 + 8*(lane>=16) + e
// element e>=8 -> k = kt*32 + 8*(lane>=16) + 16 + (e-8)
__device__ __forceinline__ vbf16 load_afrag(const __bf16* A, int tr, int kt, int lane) {
  FragBF u;
  int mrow  = tr * 16 + (lane & 15);
  int khalf = ((lane >> 4) & 1) << 3;
  const __bf16* ap = A + mrow * LDA + kt * 32 + khalf;
  u.q[0] = *(const uint4*)(ap);
  u.q[1] = *(const uint4*)(ap + 16);
  return u.v;
}

// one GEMM stage: (16x256 complex bf16 A from LDS) x (256x64 bf16 B) per wave
__device__ __forceinline__ void gemm_stage(const __bf16* Are, const __bf16* Aim,
                                           const __bf16* __restrict__ Bp,
                                           int tr, int ntBase, int lane,
                                           v8f accRe[4], v8f accIm[4]) {
#pragma unroll
  for (int kt = 0; kt < 8; ++kt) {
    vbf16 ar = load_afrag(Are, tr, kt, lane);
    vbf16 ai = load_afrag(Aim, tr, kt, lane);
#pragma unroll
    for (int ct = 0; ct < 4; ++ct) {
      vbf16 b = load_bfrag(Bp, ntBase + ct, kt, lane);
      accRe[ct] = __builtin_amdgcn_wmma_f32_16x16x32_bf16(
          false, ar, false, b, (short)0, accRe[ct], false, false);
      accIm[ct] = __builtin_amdgcn_wmma_f32_16x16x32_bf16(
          false, ai, false, b, (short)0, accIm[ct], false, false);
    }
  }
}

// --- K3: fused pointwise + 3 GEMMs + IFFT over v ---------------------------
__global__ void __launch_bounds__(256)
fno_fused(const float2* __restrict__ X,       // (512,256) fft2(x)
          const float*  __restrict__ w0,      // (256,256)
          const __bf16* __restrict__ w1p,
          const __bf16* __restrict__ l1p,
          const __bf16* __restrict__ l2p,
          const float*  __restrict__ b1,
          const float*  __restrict__ b2,
          float2* __restrict__ out) {         // (512,256,256)
  __shared__ __align__(16) unsigned char smem[65536];
  __bf16* Are = (__bf16*)smem;                         // 32 x LDA bf16 (16896B)
  __bf16* Aim = (__bf16*)(smem + 32 * LDA * 2);        // 32 x LDA bf16
  float2 (*Crow)[256] = (float2 (*)[256])smem;         // overlay: 32 x 256 float2

  const int tid    = threadIdx.x;
  const int lane   = tid & 31;
  const int w      = tid >> 5;        // wave 0..7
  const int tr     = w & 1;           // row-tile 0..1 (16 rows each)
  const int ntBase = (w >> 1) * 4;    // 4 n-tiles (64 cols) per wave
  const int r0     = blockIdx.x * 32; // 32 (b,m)-rows per workgroup

  // ---- stage 0: h = phase[m] * crelu(X[b,m] * w0[m,:]) -> bf16 A tile ----
  for (int idx = tid; idx < 32 * 256; idx += 256) {
    int row = idx >> 8, col = idx & 255;
    int r = r0 + row, bb = r >> 8, m = r & 255;
    float2 Xv = X[bb * 256 + m];
    float  wv = w0[m * 256 + col];
    float  zr = fmaxf(Xv.x * wv, 0.0f);
    float  zi = fmaxf(Xv.y * wv, 0.0f);
    float  th = (TWO_PI / 256.0f) * (float)m;
    float  sn, cs; __sincosf(th, &sn, &cs);
    Are[row * LDA + col] = (__bf16)(cs * zr - sn * zi);
    Aim[row * LDA + col] = (__bf16)(cs * zi + sn * zr);
  }
  __syncthreads();

  v8f zero = {};
  v8f accRe[4], accIm[4];

  // ---- GEMM 1: h @ w1 ----
#pragma unroll
  for (int ct = 0; ct < 4; ++ct) { accRe[ct] = zero; accIm[ct] = zero; }
  gemm_stage(Are, Aim, w1p, tr, ntBase, lane, accRe, accIm);
  __syncthreads();
#pragma unroll
  for (int ct = 0; ct < 4; ++ct) {
#pragma unroll
    for (int g = 0; g < 8; ++g) {
      int mrow = tr * 16 + ((lane < 16) ? g : g + 8);
      int n    = (ntBase + ct) * 16 + (lane & 15);
      Are[mrow * LDA + n] = (__bf16)accRe[ct][g];
      Aim[mrow * LDA + n] = (__bf16)accIm[ct][g];
    }
  }
  __syncthreads();

  // ---- GEMM 2: h @ lin1_w^T + b1, crelu ----
#pragma unroll
  for (int ct = 0; ct < 4; ++ct) { accRe[ct] = zero; accIm[ct] = zero; }
  gemm_stage(Are, Aim, l1p, tr, ntBase, lane, accRe, accIm);
  __syncthreads();
#pragma unroll
  for (int ct = 0; ct < 4; ++ct) {
    int n = (ntBase + ct) * 16 + (lane & 15);
    float bias = b1[n];
#pragma unroll
    for (int g = 0; g < 8; ++g) {
      int mrow = tr * 16 + ((lane < 16) ? g : g + 8);
      Are[mrow * LDA + n] = (__bf16)fmaxf(accRe[ct][g] + bias, 0.0f);
      Aim[mrow * LDA + n] = (__bf16)fmaxf(accIm[ct][g], 0.0f);
    }
  }
  __syncthreads();

  // ---- GEMM 3: h @ lin2_w^T + b2 -> f32 complex rows (overlay) ----
#pragma unroll
  for (int ct = 0; ct < 4; ++ct) { accRe[ct] = zero; accIm[ct] = zero; }
  gemm_stage(Are, Aim, l2p, tr, ntBase, lane, accRe, accIm);
  __syncthreads();   // everyone done reading A; safe to overlay with Crow
#pragma unroll
  for (int ct = 0; ct < 4; ++ct) {
    int n = (ntBase + ct) * 16 + (lane & 15);
    float bias = b2[n];
#pragma unroll
    for (int g = 0; g < 8; ++g) {
      int mrow = tr * 16 + ((lane < 16) ? g : g + 8);
      Crow[mrow][n] = make_float2(accRe[ct][g] + bias, accIm[ct][g]);
    }
  }
  __syncthreads();

  // ---- per-wave 256-pt inverse FFT over v, 4 rows per wave ----
  for (int rr = 0; rr < 4; ++rr) {
    float2* s = Crow[w * 4 + rr];
    for (int n = lane; n < 256; n += 32) {
      int rn = (int)(__brev((unsigned)n) >> 24);
      if (n < rn) { float2 a = s[n]; float2 b = s[rn]; s[n] = b; s[rn] = a; }
    }
    asm volatile("s_wait_dscnt 0" ::: "memory");
    for (int st = 1; st <= 8; ++st) {
      int len = 1 << st, half = len >> 1;
#pragma unroll
      for (int t = 0; t < 4; ++t) {
        int j   = lane + t * 32;                 // 128 butterflies / stage
        int blk = j >> (st - 1);
        int off = j & (half - 1);
        int i0  = blk * len + off, i1 = i0 + half;
        float ang = TWO_PI * (float)off / (float)len;   // inverse: +
        float sn, cs; __sincosf(ang, &sn, &cs);
        float2 a = s[i0];
        float2 b = cmulf(s[i1], make_float2(cs, sn));
        s[i0] = make_float2(a.x + b.x, a.y + b.y);
        s[i1] = make_float2(a.x - b.x, a.y - b.y);
      }
      asm volatile("s_wait_dscnt 0" ::: "memory");
    }
    size_t r = (size_t)(r0 + w * 4 + rr);
    for (int v = lane; v < 256; v += 32)
      out[r * 256 + v] = make_float2(s[v].x * (1.0f / 256.0f),
                                     s[v].y * (1.0f / 256.0f));
  }
}

// --- K4: in-place 256-pt IFFT along m, tiles of 16 v ----------------------
__global__ void fno_ifft_m(float2* __restrict__ O) {
  __shared__ float2 t[256 * 16];
  const int b = blockIdx.x, v0 = blockIdx.y * 16, tid = threadIdx.x;
  for (int idx = tid; idx < 4096; idx += 256) {
    int mm = idx >> 4, vv = idx & 15;
    int rm = (int)(__brev((unsigned)mm) >> 24);
    t[rm * 16 + vv] = O[((size_t)b * 256 + mm) * 256 + v0 + vv];
  }
  __syncthreads();
  for (int st = 1; st <= 8; ++st) {
    int len = 1 << st, half = len >> 1;
    for (int wk = tid; wk < 2048; wk += 256) {
      int j = wk >> 4, vv = wk & 15;
      int blk = j >> (st - 1), off = j & (half - 1);
      int i0 = (blk * len + off) * 16 + vv, i1 = i0 + half * 16;
      float ang = TWO_PI * (float)off / (float)len;
      float sn, cs; __sincosf(ang, &sn, &cs);
      float2 a = t[i0];
      float2 bb = cmulf(t[i1], make_float2(cs, sn));
      t[i0] = make_float2(a.x + bb.x, a.y + bb.y);
      t[i1] = make_float2(a.x - bb.x, a.y - bb.y);
    }
    __syncthreads();
  }
  for (int idx = tid; idx < 4096; idx += 256) {
    int mm = idx >> 4, vv = idx & 15;
    float2 v = t[mm * 16 + vv];
    O[((size_t)b * 256 + mm) * 256 + v0 + vv] =
        make_float2(v.x * (1.0f / 256.0f), v.y * (1.0f / 256.0f));
  }
}

// --- K5: in-place 512-pt IFFT along b, tiles of 8 v -----------------------
__global__ void fno_ifft_b(float2* __restrict__ O) {
  __shared__ float2 t[512 * 8];
  const int m = blockIdx.x, v0 = blockIdx.y * 8, tid = threadIdx.x;
  for (int idx = tid; idx < 4096; idx += 256) {
    int bb = idx >> 3, vv = idx & 7;
    int rb = (int)(__brev((unsigned)bb) >> 23);
    t[rb * 8 + vv] = O[((size_t)bb * 256 + m) * 256 + v0 + vv];
  }
  __syncthreads();
  for (int st = 1; st <= 9; ++st) {
    int len = 1 << st, half = len >> 1;
    for (int wk = tid; wk < 2048; wk += 256) {
      int j = wk >> 3, vv = wk & 7;
      int blk = j >> (st - 1), off = j & (half - 1);
      int i0 = (blk * len + off) * 8 + vv, i1 = i0 + half * 8;
      float ang = TWO_PI * (float)off / (float)len;
      float sn, cs; __sincosf(ang, &sn, &cs);
      float2 a = t[i0];
      float2 bb = cmulf(t[i1], make_float2(cs, sn));
      t[i0] = make_float2(a.x + bb.x, a.y + bb.y);
      t[i1] = make_float2(a.x - bb.x, a.y - bb.y);
    }
    __syncthreads();
  }
  for (int idx = tid; idx < 4096; idx += 256) {
    int bb = idx >> 3, vv = idx & 7;
    float2 v = t[bb * 8 + vv];
    O[((size_t)bb * 256 + m) * 256 + v0 + vv] =
        make_float2(v.x * (1.0f / 512.0f), v.y * (1.0f / 512.0f));
  }
}

// ---------------------------------------------------------------------------
extern "C" void kernel_launch(void* const* d_in, const int* in_sizes, int n_in,
                              void* d_out, int out_size, void* d_ws, size_t ws_size,
                              hipStream_t stream) {
  const float* x      = (const float*)d_in[0];   // (512,256)
  const float* w0     = (const float*)d_in[1];   // (256,256)
  const float* w1     = (const float*)d_in[2];   // (256,256)
  const float* lin1_w = (const float*)d_in[3];   // (256,256)
  const float* lin1_b = (const float*)d_in[4];   // (256,)
  const float* lin2_w = (const float*)d_in[5];   // (256,256)
  const float* lin2_b = (const float*)d_in[6];   // (256,)

  // workspace layout (~2.4 MB)
  char* ws = (char*)d_ws;
  float2* Xr  = (float2*)ws;                       // 512*256 complex = 1 MB
  float2* Xc  = Xr + 512 * 256;                    // 1 MB
  __bf16* w1p = (__bf16*)(Xc + 512 * 256);         // 128 KB each
  __bf16* l1p = w1p + 65536;
  __bf16* l2p = l1p + 65536;

  float2* O = (float2*)d_out;                      // (512,256,256) complex

  fno_repack  <<<(3 * 65536 + 255) / 256, 256, 0, stream>>>(w1, lin1_w, lin2_w,
                                                            w1p, l1p, l2p);
  fno_fft_rows<<<512, 128, 0, stream>>>(x, Xr);
  fno_fft_cols<<<256, 256, 0, stream>>>(Xr, Xc);
  fno_fused   <<<4096, 256, 0, stream>>>(Xc, w0, w1p, l1p, l2p,
                                         lin1_b, lin2_b, O);
  fno_ifft_m  <<<dim3(512, 16), 256, 0, stream>>>(O);
  fno_ifft_b  <<<dim3(256, 32), 256, 0, stream>>>(O);
}
